// MoE_12317966205425
// MI455X (gfx1250) — compile-verified, matrix operand order checked by reading, not run on hardware
//
#include <hip/hip_runtime.h>
#include <math.h>

// ---------------------------------------------------------------------------
// MoE capsule experts on MI455X (gfx1250, wave32):
//   3x3 conv (implicit GEMM, bf16 WMMA, f32 acc) -> bias -> squash (fused,
//   cross-wave LDS reduction) -> 1x1 conv (bf16 WMMA) -> bf16 expert tensor
//   (L2-resident, 67 MB) -> 4x top-2 gated combine (fp32 out) + aux loss.
// Interior blocks stage the input patch with the Tensor Data Mover
// (tensor_load_to_lds + s_wait_tensorcnt); border blocks use a manual
// zero-padded staging loop (SAME padding).
// ---------------------------------------------------------------------------

typedef unsigned short u16;
typedef unsigned int   u32;
typedef unsigned long long u64;
typedef __attribute__((ext_vector_type(8)))  float  v8f;
typedef __attribute__((ext_vector_type(8)))  __bf16 v8bf;
typedef __attribute__((ext_vector_type(16))) __bf16 v16bf;
typedef __attribute__((ext_vector_type(4)))  unsigned int v4u;
typedef __attribute__((ext_vector_type(4)))  int v4i;
typedef __attribute__((ext_vector_type(8)))  int v8i;

#define B_   8
#define C_   128
#define H_   64
#define W_   64
#define E_   8
#define KS3  36   // k-steps for 3x3 conv: 9*128/32
#define KS1  4    // k-steps for 1x1 conv: 128/32

// workspace layout (bytes)
#define OFF_XB   0ull                     // x in NHWC bf16:      8*64*64*128*2 = 8388608
#define OFF_WC   8388608ull               // packed 3x3 weights:  8*36*8*32*16*2 = 2359296
#define OFF_WP   10747904ull              // packed 1x1 weights:  8*4*8*32*16*2  = 262144
#define OFF_X0   11010048ull              // per-(b,c) means:     1024 floats
#define OFF_SELI 11014144ull              // top-2 indices:       4*8*2 ints
#define OFF_SELW 11014400ull              // top-2 weights:       4*8*2 floats
#define OFF_EO   11014656ull              // expert outputs bf16: 8*8*128*64*64*2 = 67108864

__device__ __forceinline__ u16 f2bf(float f) {
    u32 u = __float_as_uint(f);
    u32 r = u + 0x7FFFu + ((u >> 16) & 1u);   // round-to-nearest-even
    return (u16)(r >> 16);
}
__device__ __forceinline__ float bf2f(u16 h) {
    return __uint_as_float(((u32)h) << 16);
}

// --- x (NCHW f32) -> xb (NHWC bf16), channel-contiguous for GEMM-K loads ---
__global__ __launch_bounds__(256) void pack_x_kernel(const float* __restrict__ x,
                                                     u16* __restrict__ xb) {
    int idx = blockIdx.x * 256 + threadIdx.x;          // over NCHW
    int b = idx >> 19;                                 // 128*4096
    int r = idx & 524287;
    int c = r >> 12;
    int p = r & 4095;
    int y = p >> 6, xx = p & 63;
    xb[((((b << 6) + y) << 6) + xx) * 128 + c] = f2bf(x[idx]);
}

// --- pack 3x3 weights into per-lane WMMA B-fragment layout -----------------
// wc[e][step(36)][nchunk(8)][lane(32)][j(16)] ; lane half h=lane>>4, n=nc*16+(lane&15)
// j<8  -> k = step*32 + h*8 + j
// j>=8 -> k = step*32 + 16 + h*8 + (j-8) ; k = (ky*3+kx)*128 + c
__global__ __launch_bounds__(256) void pack_wcaps_kernel(const float* __restrict__ Wc,
                                                         u16* __restrict__ wc) {
    int idx = blockIdx.x * 256 + threadIdx.x;          // 8*36*8*32*16 = 1179648
    int e    = idx / 147456;
    int r    = idx % 147456;
    int step = r / 4096;
    int r2   = r & 4095;
    int nc   = r2 >> 9;
    int lane = (r2 >> 4) & 31;
    int j    = r2 & 15;
    int h    = lane >> 4;
    int n    = nc * 16 + (lane & 15);
    int k    = step * 32 + ((j < 8) ? (h * 8 + j) : (16 + h * 8 + (j - 8)));
    int c    = k & 127;
    int pq   = k >> 7;
    int ky   = pq / 3, kx = pq % 3;
    float w  = Wc[(((e * 128 + n) * 128 + c) * 3 + ky) * 3 + kx];
    wc[idx]  = f2bf(w);
}

// --- pack 1x1 weights, same fragment layout, k = c -------------------------
__global__ __launch_bounds__(256) void pack_wproj_kernel(const float* __restrict__ Wp,
                                                         u16* __restrict__ wp) {
    int idx = blockIdx.x * 256 + threadIdx.x;          // 8*4*8*32*16 = 131072
    int e    = idx / 16384;
    int r    = idx % 16384;
    int step = r / 4096;
    int r2   = r & 4095;
    int nc   = r2 >> 9;
    int lane = (r2 >> 4) & 31;
    int j    = r2 & 15;
    int h    = lane >> 4;
    int n    = nc * 16 + (lane & 15);
    int c    = step * 32 + ((j < 8) ? (h * 8 + j) : (16 + h * 8 + (j - 8)));
    wp[idx]  = f2bf(Wp[(e * 128 + n) * 128 + c]);
}

// --- per-(b,c) spatial mean for the gates ----------------------------------
__global__ __launch_bounds__(256) void mean_kernel(const float* __restrict__ x,
                                                   float* __restrict__ x0) {
    __shared__ float red[256];
    int bc = blockIdx.x;
    const float* p = x + (size_t)bc * 4096;
    float s = 0.f;
    for (int i = threadIdx.x; i < 4096; i += 256) s += p[i];
    red[threadIdx.x] = s;
    __syncthreads();
    for (int off = 128; off > 0; off >>= 1) {
        if (threadIdx.x < off) red[threadIdx.x] += red[threadIdx.x + off];
        __syncthreads();
    }
    if (threadIdx.x == 0) x0[bc] = red[0] * (1.f / 4096.f);
}

// --- 4 gates: softmax, top-2, renorm, usage-variance loss ------------------
__global__ __launch_bounds__(64) void gate_kernel(const float* __restrict__ x0,
                                                  const float* __restrict__ g1,
                                                  const float* __restrict__ g2,
                                                  const float* __restrict__ g3,
                                                  const float* __restrict__ g4,
                                                  int* __restrict__ sel_i,
                                                  float* __restrict__ sel_w,
                                                  float* __restrict__ loss_out) {
    __shared__ float go[8][8];
    __shared__ float usage[8];
    __shared__ float lsum;
    const float* gs[4] = {g1, g2, g3, g4};
    int t = threadIdx.x;
    if (t == 0) lsum = 0.f;
    for (int g = 0; g < 4; ++g) {
        __syncthreads();
        {   // logits: x0 (8x128) @ gate (128x8)
            int b = t >> 3, e = t & 7;
            float s = 0.f;
            const float* xr = x0 + b * 128;
            const float* gc = gs[g];
            for (int c = 0; c < 128; ++c) s += xr[c] * gc[c * 8 + e];
            go[b][e] = s;
        }
        __syncthreads();
        if (t < 8) {        // row softmax + top-2 + renormalized weights
            int b = t;
            float mx = go[b][0];
            for (int e = 1; e < 8; ++e) mx = fmaxf(mx, go[b][e]);
            float sum = 0.f;
            for (int e = 0; e < 8; ++e) { float ex = expf(go[b][e] - mx); go[b][e] = ex; sum += ex; }
            float inv = 1.f / sum;
            for (int e = 0; e < 8; ++e) go[b][e] *= inv;
            int i0 = 0; float v0 = go[b][0];
            for (int e = 1; e < 8; ++e) if (go[b][e] > v0) { v0 = go[b][e]; i0 = e; }
            int i1 = (i0 == 0) ? 1 : 0; float v1 = go[b][i1];
            for (int e = 0; e < 8; ++e) if (e != i0 && go[b][e] > v1) { v1 = go[b][e]; i1 = e; }
            float e1 = expf(v1 - v0);             // softmax over {v0, v1}, v0 is max
            float w0 = 1.f / (1.f + e1);
            float w1 = e1 / (1.f + e1);
            sel_i[(g * 8 + b) * 2 + 0] = i0;
            sel_i[(g * 8 + b) * 2 + 1] = i1;
            sel_w[(g * 8 + b) * 2 + 0] = w0;
            sel_w[(g * 8 + b) * 2 + 1] = w1;
        }
        __syncthreads();
        if (t < 8) {        // usage over batch
            float s = 0.f;
            for (int b = 0; b < 8; ++b) s += go[b][t];
            usage[t] = s;
        }
        __syncthreads();
        if (t == 0) {
            float mean = 0.f;
            for (int e = 0; e < 8; ++e) mean += usage[e];
            mean *= (1.f / 8.f);
            float var = 0.f;
            for (int e = 0; e < 8; ++e) { float d = usage[e] - mean; var += d * d; }
            var *= (1.f / 7.f);                   // ddof=1
            lsum += var / (mean * mean + 1e-10f);
        }
    }
    __syncthreads();
    if (t == 0) *loss_out = lsum;
}

// --- fused expert kernel: 3x3 WMMA GEMM -> bias -> squash -> 1x1 WMMA GEMM -
// grid: (W/16, H, E*B); block: 256 = 8 waves; wave w owns N-chunk [16w,16w+16)
__global__ __launch_bounds__(256) void expert_kernel(const u16* __restrict__ xb,
                                                     const u16* __restrict__ wc,
                                                     const u16* __restrict__ wp,
                                                     const float* __restrict__ bcaps,
                                                     const float* __restrict__ bproj,
                                                     u16* __restrict__ eo) {
    const int xt = blockIdx.x;
    const int y  = blockIdx.y;
    const int eb = blockIdx.z;
    const int e  = eb >> 3, b = eb & 7;
    const int x0p = xt << 4;
    const int tid = threadIdx.x;
    const int wv = tid >> 5, lane = tid & 31;
    const int h  = lane >> 4;       // lane half selects K sub-block (WMMA layout)
    const int nl = lane & 15;       // A row / N column within chunk
    const int m  = nl;              // A-matrix row handled by this lane

    __shared__ __align__(16) u16  As[3 * 18 * 128];   // padded 3x18x128 input patch, bf16
    __shared__ __align__(16) u16  Ut[16 * 128];       // squashed activations, bf16
    __shared__ float Sn[16];                          // per-pixel sum of squares

    // ---- stage padded input patch (SAME): rows y-1..y+1, cols x0p-1..x0p+16
    // Interior blocks: one 2D Tensor-Data-Mover DMA (tile 2304 elem x 3 rows,
    // row stride 8192 elem, 2-byte elements) issued by wave 0.
    const bool interior = (y > 0) && (y < 63) && (xt > 0) && (xt < 3);
    if (interior) {
        if (wv == 0) {
            u64 ga = (u64)(uintptr_t)xb
                   + ((u64)((((b << 6) + (y - 1)) << 6) + (x0p - 1)) << 8); // *128 ch *2 B
            u32 lds = (u32)(uintptr_t)(&As[0]);       // low 32 bits = LDS byte offset
            v4u g0;
            g0.x = 1u;                                 // D# group0: count=1 (valid)
            g0.y = lds;                                // lds_addr [63:32]
            g0.z = (u32)(ga & 0xFFFFFFFFull);          // global_addr[31:0]
            g0.w = (u32)((ga >> 32) & 0x1FFFFFFull)    // global_addr[56:32]
                 | (2u << 30);                         // type=2 ("image")
            v8i g1;
            g1[0] = (1 << 16);                         // data_size=1 (2-byte elements)
            g1[1] = 0;                                 // tensor_dim0[15:0] (dim0 = 1<<20)
            g1[2] = 0x10;                              // tensor_dim0[31:16] | tensor_dim1[15:0]
            g1[3] = (2304 << 16) | 0x10;               // tile_dim0=2304 | tensor_dim1[31:16]
            g1[4] = 3;                                 // tile_dim1=3, tile_dim2=0
            g1[5] = 8192;                              // tensor_dim0_stride = 64*128 elem
            g1[6] = 0;                                 // stride0[47:32] | stride1[15:0]
            g1[7] = 0;                                 // stride1[47:16]
            v4i gz = {0, 0, 0, 0};
#if defined(__clang_major__) && (__clang_major__ >= 23)
            v8i gz8 = {0, 0, 0, 0, 0, 0, 0, 0};
            __builtin_amdgcn_tensor_load_to_lds(g0, g1, gz, gz, gz8, 0);
#else
            __builtin_amdgcn_tensor_load_to_lds(g0, g1, gz, gz, 0);
#endif
            __builtin_amdgcn_s_wait_tensorcnt(0);
        }
    } else {
        for (int t = tid; t < 3 * 18 * 128; t += 256) {
            int c  = t & 127;
            int cl = (t >> 7) % 18;
            int ky = (t >> 7) / 18;
            int gy = y + ky - 1, gx = x0p + cl - 1;
            u16 v = 0;
            if (gy >= 0 && gy < 64 && gx >= 0 && gx < 64)
                v = xb[((((b << 6) + gy) << 6) + gx) * 128 + c];
            As[t] = v;
        }
    }
    if (tid < 16) Sn[tid] = 0.f;
    __syncthreads();

    // ---- 3x3 conv as implicit GEMM: M=16 pixels, N=16 (per wave), K=1152 ----
    v8f acc = {};
    const u16* wbase = wc + (((size_t)(e * KS3) * 8 + wv) << 9);  // *32*16
    for (int step = 0; step < KS3; ++step) {
        int kk = step >> 2;                   // (ky,kx) index
        int c0 = (step & 3) << 5;             // c sub-block
        int ky = kk / 3, kx = kk % 3;
        int rowbase = ((ky * 18) + kx + m) << 7;
        // A fragment per WMMA 16-bit layout: half0 -> K {c0..c0+7, c0+16..23}
        //                                    half1 -> K {c0+8..15, c0+24..31}
        v8bf a0 = *(const v8bf*)(As + rowbase + c0 + h * 8);
        v8bf a1 = *(const v8bf*)(As + rowbase + c0 + 16 + h * 8);
        v16bf Af = __builtin_shufflevector(a0, a1, 0,1,2,3,4,5,6,7,8,9,10,11,12,13,14,15);
        const u16* bp = wbase + ((size_t)step << 12) + (lane << 4);
        v8bf b0 = *(const v8bf*)bp;
        v8bf b1 = *(const v8bf*)(bp + 8);
        v16bf Bf = __builtin_shufflevector(b0, b1, 0,1,2,3,4,5,6,7,8,9,10,11,12,13,14,15);
        if (step + 1 < KS3)
            __builtin_prefetch(wbase + ((size_t)(step + 1) << 12) + (lane << 4), 0, 1);
        acc = __builtin_amdgcn_wmma_f32_16x16x32_bf16(false, Af, false, Bf,
                                                      (short)0, acc, false, false);
    }

    // ---- bias + squash (per-pixel channel reduction across all 8 waves) ----
    float bias = bcaps[e * 128 + wv * 16 + nl];
    float u[8], sq[8];
    #pragma unroll
    for (int i = 0; i < 8; ++i) { u[i] = acc[i] + bias; sq[i] = u[i] * u[i]; }
    // reduce across the 16 lanes of each half (xor of bits 0..3 stays in-half)
    #pragma unroll
    for (int mask = 1; mask < 16; mask <<= 1)
        #pragma unroll
        for (int i = 0; i < 8; ++i) sq[i] += __shfl_xor(sq[i], mask, 32);
    if (nl == 0) {
        #pragma unroll
        for (int i = 0; i < 8; ++i) atomicAdd(&Sn[i + h * 8], sq[i]);  // row m = i + 8h
    }
    __syncthreads();
    #pragma unroll
    for (int i = 0; i < 8; ++i) {
        int   mi = i + h * 8;
        float sn = Sn[mi];
        float sc = sn / (1.f + sn) / (sqrtf(sn) + 1e-8f);
        Ut[(mi << 7) + wv * 16 + nl] = f2bf(u[i] * sc);
    }
    __syncthreads();

    // ---- 1x1 projection GEMM: K = 128 over the squashed LDS tile ----
    v8f acc2 = {};
    const u16* wbase2 = wp + (((size_t)(e * KS1) * 8 + wv) << 9);
    for (int step = 0; step < KS1; ++step) {
        int c0 = step << 5;
        v8bf a0 = *(const v8bf*)(Ut + (m << 7) + c0 + h * 8);
        v8bf a1 = *(const v8bf*)(Ut + (m << 7) + c0 + 16 + h * 8);
        v16bf Af = __builtin_shufflevector(a0, a1, 0,1,2,3,4,5,6,7,8,9,10,11,12,13,14,15);
        const u16* bp = wbase2 + ((size_t)step << 12) + (lane << 4);
        v8bf b0 = *(const v8bf*)bp;
        v8bf b1 = *(const v8bf*)(bp + 8);
        v16bf Bf = __builtin_shufflevector(b0, b1, 0,1,2,3,4,5,6,7,8,9,10,11,12,13,14,15);
        acc2 = __builtin_amdgcn_wmma_f32_16x16x32_bf16(false, Af, false, Bf,
                                                       (short)0, acc2, false, false);
    }
    float bias2 = bproj[e * 128 + wv * 16 + nl];
    // store NCHW bf16: eo[e][b][ch][y][x0p + (i + 8h)]
    int ch = wv * 16 + nl;
    size_t base = (((size_t)((e * 8 + b) * 128 + ch)) << 12) + ((size_t)y << 6) + x0p;
    #pragma unroll
    for (int i = 0; i < 8; ++i) eo[base + i + h * 8] = f2bf(acc2[i] + bias2);
}

// --- top-2 weighted combine per gate: y = w0*E[i0] + w1*E[i1] (fp32 out) ---
__global__ __launch_bounds__(256) void combine_kernel(const u16* __restrict__ eo,
                                                      const int* __restrict__ sel_i,
                                                      const float* __restrict__ sel_w,
                                                      float* __restrict__ out) {
    int g = blockIdx.z, b = blockIdx.y;
    int i0 = sel_i[(g * 8 + b) * 2 + 0];
    int i1 = sel_i[(g * 8 + b) * 2 + 1];
    float w0 = sel_w[(g * 8 + b) * 2 + 0];
    float w1 = sel_w[(g * 8 + b) * 2 + 1];
    int t = blockIdx.x * 256 + threadIdx.x;                 // 131072 threads, 4 elems each
    size_t p = (size_t)t << 2;
    const u16* p0 = eo + (((size_t)(i0 * 8 + b)) << 19) + p;   // 128*4096 elems per (e,b)
    const u16* p1 = eo + (((size_t)(i1 * 8 + b)) << 19) + p;
    uint2 a = *(const uint2*)p0;
    uint2 c = *(const uint2*)p1;
    float4 r;
    r.x = w0 * bf2f((u16)(a.x & 0xFFFF)) + w1 * bf2f((u16)(c.x & 0xFFFF));
    r.y = w0 * bf2f((u16)(a.x >> 16))    + w1 * bf2f((u16)(c.x >> 16));
    r.z = w0 * bf2f((u16)(a.y & 0xFFFF)) + w1 * bf2f((u16)(c.y & 0xFFFF));
    r.w = w0 * bf2f((u16)(a.y >> 16))    + w1 * bf2f((u16)(c.y >> 16));
    *(float4*)(out + ((size_t)g << 22) + (((size_t)b) << 19) + p) = r;
}

extern "C" void kernel_launch(void* const* d_in, const int* in_sizes, int n_in,
                              void* d_out, int out_size, void* d_ws, size_t ws_size,
                              hipStream_t stream) {
    const float* x  = (const float*)d_in[0];
    const float* g1 = (const float*)d_in[1];
    const float* g2 = (const float*)d_in[2];
    const float* g3 = (const float*)d_in[3];
    const float* g4 = (const float*)d_in[4];
    const float* Wc = (const float*)d_in[5];
    const float* bc = (const float*)d_in[6];
    const float* Wp = (const float*)d_in[7];
    const float* bp = (const float*)d_in[8];
    float* out = (float*)d_out;

    char* ws = (char*)d_ws;
    u16*   xb    = (u16*)(ws + OFF_XB);
    u16*   wcp   = (u16*)(ws + OFF_WC);
    u16*   wpp   = (u16*)(ws + OFF_WP);
    float* x0    = (float*)(ws + OFF_X0);
    int*   sel_i = (int*)(ws + OFF_SELI);
    float* sel_w = (float*)(ws + OFF_SELW);
    u16*   eo    = (u16*)(ws + OFF_EO);

    pack_x_kernel<<<16384, 256, 0, stream>>>(x, xb);
    pack_wcaps_kernel<<<4608, 256, 0, stream>>>(Wc, wcp);
    pack_wproj_kernel<<<512, 256, 0, stream>>>(Wp, wpp);
    mean_kernel<<<1024, 256, 0, stream>>>(x, x0);
    gate_kernel<<<1, 64, 0, stream>>>(x0, g1, g2, g3, g4, sel_i, sel_w,
                                      out + (size_t)out_size - 1);

    dim3 gconv(W_ / 16, H_, E_ * B_);
    expert_kernel<<<gconv, 256, 0, stream>>>(xb, wcp, wpp, bc, bp, eo);

    dim3 gcomb(512, B_, 4);
    combine_kernel<<<gcomb, 256, 0, stream>>>(eo, sel_i, sel_w, out);
}
